// GetRep_64759516890018
// MI455X (gfx1250) — compile-verified, matrix operand order checked by reading, not run on hardware
//
#include <hip/hip_runtime.h>
#include <hip/hip_bf16.h>
#include <math.h>

typedef __attribute__((ext_vector_type(16))) _Float16 v16h;
typedef __attribute__((ext_vector_type(8)))  float    v8f;

// Fast sigmoid: prefer the gfx1250 hardware tanh (1 trans op), else exp+rcp
// (2 trans ops). Avoids the 7-op IEEE division sequence seen in round-0 asm.
__device__ __forceinline__ float fast_sigmoid(float t) {
#if __has_builtin(__builtin_amdgcn_tanhf)
    return __builtin_amdgcn_tanhf(t * 0.5f) * 0.5f + 0.5f;
#else
    return __builtin_amdgcn_rcpf(1.0f + __expf(-t));
#endif
}

// ---------------------------------------------------------------------------
// Kernel 1: partial sums of attn_maps (grid-stride, float4), deterministic.
// ---------------------------------------------------------------------------
__global__ void __launch_bounds__(256)
reduce_sum_kernel(const float* __restrict__ x, float* __restrict__ partials,
                  long long n)
{
    __shared__ float sm[256];
    const long long stride = (long long)gridDim.x * blockDim.x;
    long long i = (long long)blockIdx.x * blockDim.x + threadIdx.x;
    const float4* x4 = (const float4*)x;
    const long long n4 = n >> 2;   // n is a multiple of 4 (512*8*16384)
    float s = 0.f;
    for (long long k = i; k < n4; k += stride) {
        float4 v = x4[k];
        s += v.x + v.y + v.z + v.w;
    }
    sm[threadIdx.x] = s;
    __syncthreads();
    for (int off = 128; off > 0; off >>= 1) {
        if (threadIdx.x < off) sm[threadIdx.x] += sm[threadIdx.x + off];
        __syncthreads();
    }
    if (threadIdx.x == 0) partials[blockIdx.x] = sm[0];
}

// ---------------------------------------------------------------------------
// Kernel 2: reduce 1024 partials -> global mean.
// ---------------------------------------------------------------------------
__global__ void __launch_bounds__(1024)
finalize_mean_kernel(const float* __restrict__ partials,
                     float* __restrict__ mean_out, float inv_n)
{
    __shared__ float sm[1024];
    sm[threadIdx.x] = partials[threadIdx.x];
    __syncthreads();
    for (int off = 512; off > 0; off >>= 1) {
        if (threadIdx.x < off) sm[threadIdx.x] += sm[threadIdx.x + off];
        __syncthreads();
    }
    if (threadIdx.x == 0) mean_out[0] = sm[0] * inv_n;
}

// ---------------------------------------------------------------------------
// Kernel 3: one workgroup (8 wave32) per batch.
//   Each wave covers a 2048-wide L-chunk; a 16x32 f16 A-tile packs rows 0-7 =
//   channels over the chunk's first 1024, rows 8-15 = channels over the second
//   1024. v_wmma_f32_16x16x32_f16 with A==B gives D = A*A^T; the 8x8 Gram is
//   D[0:8,0:8] + D[8:16,8:16]. Row sums accumulated in f32 alongside.
//   union(i,j) = s_i + s_j - inter(i,j); st[b] = sum_tril |inter/union - 100*tm|.
// ---------------------------------------------------------------------------
__global__ void __launch_bounds__(256)
batch_iou_kernel(const float* __restrict__ attn, const float* __restrict__ tm,
                 const float* __restrict__ bias, const float* __restrict__ meanp,
                 float* __restrict__ st_out)
{
    constexpr int C = 8;
    constexpr int L = 16384;

    const int b      = blockIdx.x;
    const int tid    = threadIdx.x;
    const int wave   = tid >> 5;
    const int lane   = tid & 31;
    const int m      = lane & 15;        // WMMA A-row (M) handled by this lane
    const int ch     = m & 7;            // channel for this row
    const int laneHi = lane >> 4;        // lanes 16-31 hold K in {8..15, 24..31}

    const float mean = meanp[0];
    const float bch  = bias[ch];

    const float* rowp = attn + ((size_t)b * C + ch) * L;
    // wave chunk base + (row>=8 ? second 1024 : first 1024) + lane-half K offset
    const int lbase = wave * 2048 + ((m >> 3) ? 1024 : 0) + (laneHi ? 8 : 0);

    v8f  acc = {};
    float rs = 0.f;

    for (int kb = 0; kb < 1024; kb += 32) {
        const float* p = rowp + lbase + kb;      // 32B aligned
        float4 v0 = *(const float4*)(p);
        float4 v1 = *(const float4*)(p + 4);
        float4 v2 = *(const float4*)(p + 16);
        float4 v3 = *(const float4*)(p + 20);
        float f[16] = { v0.x, v0.y, v0.z, v0.w,  v1.x, v1.y, v1.z, v1.w,
                        v2.x, v2.y, v2.z, v2.w,  v3.x, v3.y, v3.z, v3.w };
        v16h a = {};
#pragma unroll
        for (int e = 0; e < 16; ++e) {
            float t = 500.f * (f[e] - mean) + bch;  // 5*(100*(a-mean)) + bias
            float s = fast_sigmoid(t);              // b5 = sigmoid(t)
            rs  += s;
            a[e] = (_Float16)s;
        }
        // D += A * A^T  (same registers as A and B operands)
        acc = __builtin_amdgcn_wmma_f32_16x16x32_f16(
                  false, a, false, a, (short)0, acc, false, false);
    }

    __shared__ float sD[8][256];   // per-wave 16x16 D tiles
    __shared__ float sRS[8][32];   // per-lane row-sum partials
    __shared__ float sG[64];       // 8x8 Gram
    __shared__ float sS[8];        // row sums
    __shared__ float sAbs[64];     // |sim - 100*tm| terms

    {
        // C/D layout: VGPR r -> M = r (+8 for lanes 16-31), N = lane & 15
        const int N = lane & 15;
#pragma unroll
        for (int r = 0; r < 8; ++r) {
            const int M = r + (laneHi ? 8 : 0);
            sD[wave][M * 16 + N] = acc[r];
        }
        sRS[wave][lane] = rs;
    }
    __syncthreads();

    if (tid < 64) {
        const int i = tid >> 3, j = tid & 7;
        float g = 0.f;
#pragma unroll
        for (int w = 0; w < 8; ++w)
            g += sD[w][i * 16 + j] + sD[w][(i + 8) * 16 + (j + 8)];
        sG[tid] = g;
    }
    if (tid < 8) {
        // lanes {ch, ch+8, ch+16, ch+24} all map to channel ch
        float s = 0.f;
#pragma unroll
        for (int w = 0; w < 8; ++w)
            s += sRS[w][tid] + sRS[w][tid + 8] + sRS[w][tid + 16] + sRS[w][tid + 24];
        sS[tid] = s;
    }
    __syncthreads();

    if (tid < 64) {
        const int i = tid >> 3, j = tid & 7;
        float v = 0.f;
        if (i >= j) {                          // lower triangle incl. diagonal
            const float inter = sG[tid];
            const float uni   = sS[i] + sS[j] - inter;   // L - (1-bi)(1-bj)
            const float sim   = inter / uni;
            v = fabsf(sim - 100.f * tm[(size_t)b * 64 + tid]);
        }
        sAbs[tid] = v;
    }
    __syncthreads();

    if (tid == 0) {
        float s = 0.f;
        for (int k = 0; k < 64; ++k) s += sAbs[k];
        st_out[b] = s;
    }
}

// ---------------------------------------------------------------------------
// Kernel 4: BatchNorm1d(1, affine=False, training) over B=512, then 1->3 linear.
// ---------------------------------------------------------------------------
__global__ void __launch_bounds__(512)
bn_fc_kernel(const float* __restrict__ st, const float* __restrict__ w,
             const float* __restrict__ bfc, float* __restrict__ out)
{
    __shared__ float ssum[512];
    __shared__ float ssq[512];
    const int t = threadIdx.x;
    const float v = st[t];
    ssum[t] = v;
    ssq[t]  = v * v;
    __syncthreads();
    for (int off = 256; off > 0; off >>= 1) {
        if (t < off) { ssum[t] += ssum[t + off]; ssq[t] += ssq[t + off]; }
        __syncthreads();
    }
    const float mu  = ssum[0] * (1.0f / 512.0f);
    const float var = ssq[0] * (1.0f / 512.0f) - mu * mu;   // biased variance
    const float stn = (v - mu) * rsqrtf(var + 1e-5f);
    out[t * 3 + 0] = stn * w[0] + bfc[0];
    out[t * 3 + 1] = stn * w[1] + bfc[1];
    out[t * 3 + 2] = stn * w[2] + bfc[2];
}

// ---------------------------------------------------------------------------
extern "C" void kernel_launch(void* const* d_in, const int* in_sizes, int n_in,
                              void* d_out, int out_size, void* d_ws, size_t ws_size,
                              hipStream_t stream)
{
    const float* attn = (const float*)d_in[0];   // (512, 8, 16384) f32
    const float* tm   = (const float*)d_in[1];   // (512, 8, 8) f32
    const float* bias = (const float*)d_in[2];   // (8,) f32
    const float* wfc  = (const float*)d_in[3];   // (3, 1) f32
    const float* bfc  = (const float*)d_in[4];   // (3,) f32
    float* out = (float*)d_out;                  // (512, 3) f32

    float* ws       = (float*)d_ws;
    float* meanbuf  = ws;            // [1]
    float* partials = ws + 1;        // [1024]
    float* stbuf    = ws + 1 + 1024; // [512]

    const long long n = (long long)in_sizes[0]; // 512*8*16384

    reduce_sum_kernel  <<<1024, 256, 0, stream>>>(attn, partials, n);
    finalize_mean_kernel<<<1, 1024, 0, stream>>>(partials, meanbuf, 1.0f / (float)n);
    batch_iou_kernel   <<<512, 256, 0, stream>>>(attn, tm, bias, meanbuf, stbuf);
    bn_fc_kernel       <<<1, 512, 0, stream>>>(stbuf, wfc, bfc, out);
}
